// GINModel_66159676227906
// MI455X (gfx1250) — matrix operand order repaired and checked by reading, not compiled, not run on hardware
//
#include <hip/hip_runtime.h>
#include <hip/hip_bf16.h>

// ---------------------------------------------------------------------------
// GIN model for MI455X (gfx1250): scatter-add aggregation + WMMA bf16 MLPs.
// Round 2: unconditional vectorized fragment loads + pre-packed B fragments.
// ---------------------------------------------------------------------------

#define GIN_N 100000
#define GIN_E 3200000
#define GIN_F 13
#define GIN_FP 32      // layer-1 K padded to one WMMA k-step
#define GIN_H 64
#define GIN_G 512
#define GIN_C 2
#define BN_EPS 1e-5f

typedef __attribute__((ext_vector_type(16))) __bf16 v16bf;
typedef __attribute__((ext_vector_type(8)))  float  v8f;

// ---------------------------------------------------------------------------
// Elementwise helpers
// ---------------------------------------------------------------------------
__global__ void k_zero_f32(float* __restrict__ p, long long n) {
    long long stride = (long long)gridDim.x * blockDim.x;
    for (long long i = (long long)blockIdx.x * blockDim.x + threadIdx.x; i < n; i += stride)
        p[i] = 0.0f;
}

__global__ void k_copy_f32(const float* __restrict__ in, float* __restrict__ out, long long n) {
    long long stride = (long long)gridDim.x * blockDim.x;
    for (long long i = (long long)blockIdx.x * blockDim.x + threadIdx.x; i < n; i += stride)
        out[i] = in[i];
}

// copy x[N,13] into zero-padded agg[N,32]
__global__ void k_pad_copy13(const float* __restrict__ x, float* __restrict__ agg) {
    long long total = (long long)GIN_N * GIN_F;
    long long stride = (long long)gridDim.x * blockDim.x;
    for (long long i = (long long)blockIdx.x * blockDim.x + threadIdx.x; i < total; i += stride) {
        int n = (int)(i / GIN_F);
        int f = (int)(i % GIN_F);
        agg[(size_t)n * GIN_FP + f] = x[i];
    }
}

// ---------------------------------------------------------------------------
// Edge scatter-add: agg[dst] += h[src].
// ---------------------------------------------------------------------------
__global__ void k_scatter13(const float* __restrict__ x, const int* __restrict__ src,
                            const int* __restrict__ dst, float* __restrict__ agg) {
    long long total = (long long)GIN_E * 16;
    long long stride = (long long)gridDim.x * blockDim.x;
    for (long long i = (long long)blockIdx.x * blockDim.x + threadIdx.x; i < total; i += stride) {
        int e = (int)(i >> 4);
        int q = (int)(i & 15);
        if (q < GIN_F) {
            int s = src[e], d = dst[e];
            atomicAdd(&agg[(size_t)d * GIN_FP + q], x[(size_t)s * GIN_F + q]);
        }
    }
}

__global__ void k_scatter64(const float* __restrict__ h, const int* __restrict__ src,
                            const int* __restrict__ dst, float* __restrict__ agg) {
    long long total = (long long)GIN_E * 16;
    long long stride = (long long)gridDim.x * blockDim.x;
    for (long long i = (long long)blockIdx.x * blockDim.x + threadIdx.x; i < total; i += stride) {
        int e = (int)(i >> 4);
        int q = (int)(i & 15);           // 16 lanes x 4 floats = 64 features
        int s = src[e], d = dst[e];
        const float4 v = *(const float4*)(h + (size_t)s * GIN_H + q * 4);
        float* p = agg + (size_t)d * GIN_H + q * 4;
        atomicAdd(p + 0, v.x);
        atomicAdd(p + 1, v.y);
        atomicAdd(p + 2, v.z);
        atomicAdd(p + 3, v.w);
    }
}

// ---------------------------------------------------------------------------
// Pack W[fin,64] (f32, row-major) into per-lane WMMA B fragments (bf16).
// Fragment layout (CDNA5 V_WMMA_F32_16X16X32_BF16, wave32):
//   B 32x16 (KxN): lane = half*16+idx; col = idx; elem e -> K = kb + half*8 + e (e<8), kb + 16 + half*8 + (e-8) (e>=8)
// Packed as Bpk[t][ct][lane][e]  (t = kb/32, ct = column tile 0..3),
// so each lane reads its 16 bf16 (32 bytes) contiguously.
// ---------------------------------------------------------------------------
__global__ void k_pack_w(const float* __restrict__ W, int fin, int kpad,
                         __bf16* __restrict__ out) {
    int total = (kpad / 32) * 4 * 32 * 16;
    int i = blockIdx.x * blockDim.x + threadIdx.x;
    if (i >= total) return;
    int e    = i & 15;
    int lane = (i >> 4) & 31;
    int ct   = (i >> 9) & 3;
    int t    = i >> 11;
    int half = lane >> 4;
    int idx  = lane & 15;
    int k = t * 32 + (e < 8 ? half * 8 + e : 16 + half * 8 + (e - 8));
    int c = ct * 16 + idx;
    float f = (k < fin) ? W[(size_t)k * GIN_H + c] : 0.0f;
    out[i] = (__bf16)f;
}

// ---------------------------------------------------------------------------
// Fused Linear(+BN)+ReLU:  out[N,64] = relu(bn(A[N,kpad] @ W + bias))
// A is f32 row-major with ldA == kpad (32 or 64), physically zero-padded.
// Block = 256 threads = 8 waves; wave w owns rows [blk*128 + w*16, +16),
// 4 accumulator tiles covering cols 0..63, K-loop step 32 with
// v_wmma_f32_16x16x32_bf16.  No divergence before/around the WMMAs.
// ---------------------------------------------------------------------------
__global__ __launch_bounds__(256) void k_gemm_mlp(
    const float* __restrict__ A, int kpad,
    const __bf16* __restrict__ Bpk, const float* __restrict__ bias,
    const float* __restrict__ bn_g, const float* __restrict__ bn_b,
    const float* __restrict__ bn_m, const float* __restrict__ bn_v,
    int use_bn, float* __restrict__ out, int nrows)
{
    const int wave = threadIdx.x >> 5;
    const int lane = threadIdx.x & 31;
    const int half = lane >> 4;
    const int idx  = lane & 15;

    const int rowBase = blockIdx.x * 128 + wave * 16;
    // clamp instead of predicating: loads always legal, EXEC stays all-ones
    int arow = rowBase + idx;
    arow = (arow < nrows) ? arow : (nrows - 1);
    const float* rp = A + (size_t)arow * kpad;

    v8f acc[4];
#pragma unroll
    for (int ct = 0; ct < 4; ++ct) acc[ct] = (v8f){};

    const int nsteps = kpad >> 5;
    for (int t = 0; t < nsteps; ++t) {
        // A fragment: elems 0..7 -> K = k0..k0+7, elems 8..15 -> K = k0+16..k0+23
        const int k0 = t * 32 + half * 8;
        const float4 a0 = *(const float4*)(rp + k0);
        const float4 a1 = *(const float4*)(rp + k0 + 4);
        const float4 a2 = *(const float4*)(rp + k0 + 16);
        const float4 a3 = *(const float4*)(rp + k0 + 20);
        v16bf a;
        a[0]  = (__bf16)a0.x; a[1]  = (__bf16)a0.y; a[2]  = (__bf16)a0.z; a[3]  = (__bf16)a0.w;
        a[4]  = (__bf16)a1.x; a[5]  = (__bf16)a1.y; a[6]  = (__bf16)a1.z; a[7]  = (__bf16)a1.w;
        a[8]  = (__bf16)a2.x; a[9]  = (__bf16)a2.y; a[10] = (__bf16)a2.z; a[11] = (__bf16)a2.w;
        a[12] = (__bf16)a3.x; a[13] = (__bf16)a3.y; a[14] = (__bf16)a3.z; a[15] = (__bf16)a3.w;

#pragma unroll
        for (int ct = 0; ct < 4; ++ct) {
            const v16bf b = *(const v16bf*)(Bpk + (((size_t)t * 4 + ct) * 32 + lane) * 16);
            acc[ct] = __builtin_amdgcn_wmma_f32_16x16x32_bf16(
                false, a, false, b, (short)0, acc[ct], false, false);
        }
    }

#pragma unroll
    for (int ct = 0; ct < 4; ++ct) {
        const int col = ct * 16 + idx;
        const float bcol = bias[col];
        float scale = 1.0f, shift = 0.0f;
        if (use_bn) {
            scale = rsqrtf(bn_v[col] + BN_EPS) * bn_g[col];
            shift = bn_b[col] - bn_m[col] * scale;
        }
#pragma unroll
        for (int r = 0; r < 8; ++r) {
            const int row = rowBase + half * 8 + r;
            if (row < nrows) {
                float val = acc[ct][r] + bcol;
                if (use_bn) val = val * scale + shift;
                val = fmaxf(val, 0.0f);
                out[(size_t)row * GIN_H + col] = val;
            }
        }
    }
}

// ---------------------------------------------------------------------------
// Graph pooling: pool[batch[n], off + f] += h[n, f]
// ---------------------------------------------------------------------------
__global__ void k_pool(const float* __restrict__ h, const int* __restrict__ batch,
                       float* __restrict__ pool, int off) {
    long long total = (long long)GIN_N * GIN_H;
    long long stride = (long long)gridDim.x * blockDim.x;
    for (long long i = (long long)blockIdx.x * blockDim.x + threadIdx.x; i < total; i += stride) {
        int n = (int)(i >> 6);
        int f = (int)(i & 63);
        atomicAdd(&pool[(size_t)batch[n] * (3 * GIN_H) + off + f], h[i]);
    }
}

// ---------------------------------------------------------------------------
// FC1: [G,192] @ [192,192] + b, ReLU  (one thread per output element)
// ---------------------------------------------------------------------------
__global__ void k_fc1(const float* __restrict__ in, const float* __restrict__ W,
                      const float* __restrict__ b, float* __restrict__ out) {
    int i = blockIdx.x * blockDim.x + threadIdx.x;
    const int D = 3 * GIN_H;
    if (i >= GIN_G * D) return;
    int g = i / D, j = i % D;
    const float* rp = in + (size_t)g * D;
    float s = b[j];
    for (int k = 0; k < D; ++k) s += rp[k] * W[(size_t)k * D + j];
    out[i] = fmaxf(s, 0.0f);
}

// ---------------------------------------------------------------------------
// FC2 + log_softmax (C=2): one thread per graph
// ---------------------------------------------------------------------------
__global__ void k_fc2_logsoftmax(const float* __restrict__ in, const float* __restrict__ W,
                                 const float* __restrict__ b, float* __restrict__ out) {
    int g = blockIdx.x * blockDim.x + threadIdx.x;
    if (g >= GIN_G) return;
    const int D = 3 * GIN_H;
    const float* rp = in + (size_t)g * D;
    float z0 = b[0], z1 = b[1];
    for (int k = 0; k < D; ++k) {
        float v = rp[k];
        z0 += v * W[(size_t)k * GIN_C + 0];
        z1 += v * W[(size_t)k * GIN_C + 1];
    }
    float mx  = fmaxf(z0, z1);
    float lse = mx + logf(expf(z0 - mx) + expf(z1 - mx));
    out[(size_t)g * GIN_C + 0] = z0 - lse;
    out[(size_t)g * GIN_C + 1] = z1 - lse;
}

// ---------------------------------------------------------------------------
// Host launcher
// ---------------------------------------------------------------------------
static inline unsigned grid_for(long long total, int bs) {
    long long blocks = (total + bs - 1) / bs;
    if (blocks > (1LL << 20)) blocks = 1LL << 20;
    if (blocks < 1) blocks = 1;
    return (unsigned)blocks;
}

extern "C" void kernel_launch(void* const* d_in, const int* in_sizes, int n_in,
                              void* d_out, int out_size, void* d_ws, size_t ws_size,
                              hipStream_t stream) {
    (void)in_sizes; (void)n_in; (void)out_size; (void)ws_size;

    const float* x     = (const float*)d_in[0];
    const int*   src   = (const int*)d_in[1];
    const int*   dst   = (const int*)d_in[2];
    const int*   batch = (const int*)d_in[3];

    // conv params: base = 4 + 8*layer : W1,b1,g,be,m,v,W2,b2
    const float* cp[3][8];
    for (int l = 0; l < 3; ++l)
        for (int j = 0; j < 8; ++j)
            cp[l][j] = (const float*)d_in[4 + 8 * l + j];

    const float* fc1_W = (const float*)d_in[28];
    const float* fc1_b = (const float*)d_in[29];
    const float* fc2_W = (const float*)d_in[30];
    const float* fc2_b = (const float*)d_in[31];

    // workspace layout
    float* agg  = (float*)d_ws;                        // N*64 (layer1 uses N*32 padded)
    float* mid  = agg  + (size_t)GIN_N * GIN_H;        // N*64
    float* hA   = mid  + (size_t)GIN_N * GIN_H;        // N*64
    float* hB   = hA   + (size_t)GIN_N * GIN_H;        // N*64
    float* pool = hB   + (size_t)GIN_N * GIN_H;        // G*192
    float* fc1o = pool + (size_t)GIN_G * 3 * GIN_H;    // G*192
    // packed-B slots (bf16, 4096 elems = 8KB each, 32B aligned)
    __bf16* Bpk = (__bf16*)(fc1o + (size_t)GIN_G * 3 * GIN_H);
    __bf16* Bp[6];
    for (int s = 0; s < 6; ++s) Bp[s] = Bpk + (size_t)s * 4096;

    const int BS = 256;
    const unsigned gemm_blocks = (GIN_N + 127) / 128;

    // --- weight fragment packing (tiny) ---
    // slot 0: layer1 W1 (13 -> kpad 32); slots 1..5: 64x64 weights
    k_pack_w<<<grid_for(2048, BS), BS, 0, stream>>>(cp[0][0], GIN_F, GIN_FP, Bp[0]);
    k_pack_w<<<grid_for(4096, BS), BS, 0, stream>>>(cp[0][6], GIN_H, GIN_H, Bp[1]);
    k_pack_w<<<grid_for(4096, BS), BS, 0, stream>>>(cp[1][0], GIN_H, GIN_H, Bp[2]);
    k_pack_w<<<grid_for(4096, BS), BS, 0, stream>>>(cp[1][6], GIN_H, GIN_H, Bp[3]);
    k_pack_w<<<grid_for(4096, BS), BS, 0, stream>>>(cp[2][0], GIN_H, GIN_H, Bp[4]);
    k_pack_w<<<grid_for(4096, BS), BS, 0, stream>>>(cp[2][6], GIN_H, GIN_H, Bp[5]);

    // zero the pooled sums
    k_zero_f32<<<grid_for((long long)GIN_G * 3 * GIN_H, BS), BS, 0, stream>>>(
        pool, (long long)GIN_G * 3 * GIN_H);

    // ---------------- layer 1 (K = 13 padded to 32) ----------------
    k_zero_f32<<<grid_for((long long)GIN_N * GIN_FP, BS), BS, 0, stream>>>(
        agg, (long long)GIN_N * GIN_FP);
    k_pad_copy13<<<grid_for((long long)GIN_N * GIN_F, BS), BS, 0, stream>>>(x, agg);
    k_scatter13<<<grid_for((long long)GIN_E * 16, BS), BS, 0, stream>>>(x, src, dst, agg);
    k_gemm_mlp<<<gemm_blocks, BS, 0, stream>>>(
        agg, GIN_FP, Bp[0], cp[0][1],
        cp[0][2], cp[0][3], cp[0][4], cp[0][5], 1, mid, GIN_N);
    k_gemm_mlp<<<gemm_blocks, BS, 0, stream>>>(
        mid, GIN_H, Bp[1], cp[0][7],
        cp[0][7], cp[0][7], cp[0][7], cp[0][7], 0, hA, GIN_N);
    k_pool<<<grid_for((long long)GIN_N * GIN_H, BS), BS, 0, stream>>>(hA, batch, pool, 0);

    // ---------------- layer 2 (K = 64) ----------------
    k_copy_f32<<<grid_for((long long)GIN_N * GIN_H, BS), BS, 0, stream>>>(
        hA, agg, (long long)GIN_N * GIN_H);
    k_scatter64<<<grid_for((long long)GIN_E * 16, BS), BS, 0, stream>>>(hA, src, dst, agg);
    k_gemm_mlp<<<gemm_blocks, BS, 0, stream>>>(
        agg, GIN_H, Bp[2], cp[1][1],
        cp[1][2], cp[1][3], cp[1][4], cp[1][5], 1, mid, GIN_N);
    k_gemm_mlp<<<gemm_blocks, BS, 0, stream>>>(
        mid, GIN_H, Bp[3], cp[1][7],
        cp[1][7], cp[1][7], cp[1][7], cp[1][7], 0, hB, GIN_N);
    k_pool<<<grid_for((long long)GIN_N * GIN_H, BS), BS, 0, stream>>>(hB, batch, pool, GIN_H);

    // ---------------- layer 3 (K = 64) ----------------
    k_copy_f32<<<grid_for((long long)GIN_N * GIN_H, BS), BS, 0, stream>>>(
        hB, agg, (long long)GIN_N * GIN_H);
    k_scatter64<<<grid_for((long long)GIN_E * 16, BS), BS, 0, stream>>>(hB, src, dst, agg);
    k_gemm_mlp<<<gemm_blocks, BS, 0, stream>>>(
        agg, GIN_H, Bp[4], cp[2][1],
        cp[2][2], cp[2][3], cp[2][4], cp[2][5], 1, mid, GIN_N);
    k_gemm_mlp<<<gemm_blocks, BS, 0, stream>>>(
        mid, GIN_H, Bp[5], cp[2][7],
        cp[2][7], cp[2][7], cp[2][7], cp[2][7], 0, hA, GIN_N);
    k_pool<<<grid_for((long long)GIN_N * GIN_H, BS), BS, 0, stream>>>(hA, batch, pool, 2 * GIN_H);

    // ---------------- readout head ----------------
    k_fc1<<<grid_for((long long)GIN_G * 3 * GIN_H, BS), BS, 0, stream>>>(pool, fc1_W, fc1_b, fc1o);
    k_fc2_logsoftmax<<<grid_for(GIN_G, BS), BS, 0, stream>>>(fc1o, fc2_W, fc2_b, (float*)d_out);
}
